// EDFFN_MY_8443905704006
// MI455X (gfx1250) — matrix operand order repaired and checked by reading, not compiled, not run on hardware
//
#include <hip/hip_runtime.h>
#include <hip/hip_bf16.h>
#include <math.h>

// ---------------- problem constants ----------------
#define DIMC   64
#define HID    256
#define BATCH  8
#define IMGH   256
#define IMGW   256
#define HWPIX  (IMGH*IMGW)                 // 65536
#define NTOT   ((size_t)BATCH*DIMC*HWPIX)  // 33,554,432
#define NPIX_PER_CH ((float)(BATCH*HWPIX)) // 524288 per channel for BN stats

#define TILE   16
#define HALOW  18          // TILE + 2
#define HPX    (HALOW*HALOW)   // 324
#define NPAD   336         // 21 * 16, padded pixel count for GEMM1 N
#define KCH    64          // channel-chunk (pairs): x1 chunk 64 + x2 chunk 64 = 128 rows

typedef _Float16 h16;
typedef __attribute__((ext_vector_type(8)))  _Float16 v8h;
typedef __attribute__((ext_vector_type(16))) _Float16 v16h;
typedef __attribute__((ext_vector_type(8)))  float    v8f;

// ---------------- LDS layout (dynamic, bytes) ----------------
// xlds   f16 [336][64]   43008   activations (pixel-major, ch contiguous)
// y1     f16 [336][128]  86016   conv1 chunk out (pixel-major) | fp32 staging | fp32 outls[64][256]
// g      f16 [256][64]   32768   gated chunk (pixel-major)
// wout   f16 [64][256]   32768   w_out as f16
// win    f16 [128][64]   16384   w_in chunk as f16
// wdw    f32 [128][9]     4608
// bin    f32 [128]         512
// bdw    f32 [128]         512
// bout   f32 [64]          256
// mask   f32 [64][40]    10240
// trig   f32 [16]           64
#define OFF_X     0
#define OFF_Y1    43008
#define OFF_G     129024
#define OFF_WOUT  161792
#define OFF_WIN   194560
#define OFF_WDW   210944
#define OFF_BIN   215552
#define OFF_BDW   216064
#define OFF_BOUT  216576
#define OFF_MASK  216832
#define OFF_TRIG  227072
#define K1_SMEM   227136

__device__ __forceinline__ float gelu_exact(float x) {
    return 0.5f * x * (1.0f + erff(x * 0.70710678118654752f));
}

__device__ __forceinline__ v8f wmma_f16f32(v16h a, v16h b, v8f c) {
    return __builtin_amdgcn_wmma_f32_16x16x32_f16(false, a, false, b, (short)0, c,
                                                  false, false);
}

// A-matrix (16x32 f16) lane load per ISA 7.12.2:
//   lanes 0-15 : row=lane,    v0..3 = K 0..7,  v4..7 = K 16..23
//   lanes16-31 : row=lane-16, v0..3 = K 8..15, v4..7 = K 24..31
__device__ __forceinline__ v16h load_A(const h16* base, int row, int rstride,
                                       int k0, int lane) {
    int khalf = lane >> 4;
    const v8h* p0 = (const v8h*)(base + (size_t)row * rstride + k0 + khalf * 8);
    const v8h* p1 = (const v8h*)(base + (size_t)row * rstride + k0 + 16 + khalf * 8);
    v8h lo = *p0, hi = *p1;
    v16h a;
#pragma unroll
    for (int i = 0; i < 8; ++i) { a[i] = lo[i]; a[8 + i] = hi[i]; }
    return a;
}

// B-matrix (32x16 f16) lane load:
//   lanes 0-15: col=lane, v0..3 = K 0..7, v4..7 = K 16..23 ; lanes16-31: col=lane-16, K 8..15, 24..31
// act is pixel-major with `cstride` contiguous channels per pixel.
__device__ __forceinline__ v16h load_B(const h16* act, int px, int cstride,
                                       int k0, int lane) {
    int khalf = lane >> 4;
    const v8h* p0 = (const v8h*)(act + (size_t)px * cstride + k0 + khalf * 8);
    v8h lo = p0[0], hi = p0[2];
    v16h b;
#pragma unroll
    for (int i = 0; i < 8; ++i) { b[i] = lo[i]; b[8 + i] = hi[i]; }
    return b;
}

// =====================================================================
// Kernel 1: fused conv1x1_in -> dwconv3x3 -> gelu-gate -> conv1x1_out
//           -> 8x8 patch rFFT*mask*irFFT -> +residual  => ws_y
// One workgroup (256 thr = 8 waves) per 16x16 pixel tile per batch.
// =====================================================================
__global__ __launch_bounds__(256, 1)
void k_edffn(const float* __restrict__ x,    const float* __restrict__ w_in,
             const float* __restrict__ b_in, const float* __restrict__ w_dw,
             const float* __restrict__ b_dw, const float* __restrict__ fmask,
             const float* __restrict__ w_out,const float* __restrict__ b_out,
             float* __restrict__ ws_y) {
    extern __shared__ char smem[];
    h16*   xl    = (h16*)  (smem + OFF_X);
    h16*   y1l   = (h16*)  (smem + OFF_Y1);
    h16*   gl    = (h16*)  (smem + OFF_G);
    h16*   woutl = (h16*)  (smem + OFF_WOUT);
    h16*   winl  = (h16*)  (smem + OFF_WIN);
    float* wdwl  = (float*)(smem + OFF_WDW);
    float* binl  = (float*)(smem + OFF_BIN);
    float* bdwl  = (float*)(smem + OFF_BDW);
    float* boutl = (float*)(smem + OFF_BOUT);
    float* maskl = (float*)(smem + OFF_MASK);
    float* trig  = (float*)(smem + OFF_TRIG);

    const int tid  = threadIdx.x;
    const int lane = tid & 31;
    const int wave = tid >> 5;
    const int tx0  = blockIdx.x * TILE;
    const int ty0  = blockIdx.y * TILE;
    const int b    = blockIdx.z;
    const float* xb = x + (size_t)b * DIMC * HWPIX;

    // ---- persistent LDS fills ----
    for (int i = tid; i < 64 * 256; i += 256) woutl[i] = (h16)w_out[i];
    for (int i = tid; i < 64;       i += 256) boutl[i] = b_out[i];
    for (int i = tid; i < 64 * 40;  i += 256) maskl[i] = fmask[i];
    if (tid < 8) {
        const float c8v[8] = {1.f, 0.70710678f, 0.f, -0.70710678f,
                              -1.f,-0.70710678f, 0.f,  0.70710678f};
        const float s8v[8] = {0.f, 0.70710678f, 1.f,  0.70710678f,
                              0.f,-0.70710678f,-1.f, -0.70710678f};
        trig[tid] = c8v[tid]; trig[8 + tid] = s8v[tid];
    }

    // ---- async copy x halo tile (fp32) into staging (reuses y1 region) ----
    float* stag = (float*)(smem + OFF_Y1);
    for (int i = tid; i < HPX * DIMC; i += 256) {
        int c  = i & 63;
        int hp = i >> 6;
        int gy = ty0 - 1 + hp / HALOW;
        int gx = tx0 - 1 + hp % HALOW;
        unsigned ldsaddr = (unsigned)(size_t)(void*)(stag + i);
        if (gy >= 0 && gy < IMGH && gx >= 0 && gx < IMGW) {
            unsigned long long ga =
                (unsigned long long)(const void*)(xb + (size_t)c * HWPIX + gy * IMGW + gx);
            asm volatile("global_load_async_to_lds_b32 %0, %1, off"
                         :: "v"(ldsaddr), "v"(ga) : "memory");
        } else {
            stag[i] = 0.0f;
        }
    }
    asm volatile("s_wait_asynccnt 0" ::: "memory");
    __syncthreads();

    // fp32 staging -> f16 activations (pixel-major), zero-pad px 324..335
    for (int i = tid; i < HPX * DIMC; i += 256) xl[i] = (h16)stag[i];
    for (int i = HPX * DIMC + tid; i < NPAD * DIMC; i += 256) xl[i] = (h16)0.0f;
    __syncthreads();

    // conv1x1_out accumulators: 64 x 256px = 4 Mtiles x 16 Ntiles, 8 per wave
    v8f acc[8];
#pragma unroll
    for (int i = 0; i < 8; ++i) acc[i] = (v8f){0.f,0.f,0.f,0.f,0.f,0.f,0.f,0.f};
    const int mi2   = wave >> 1;        // output-channel tile for GEMM2
    const int nbase = (wave & 1) * 8;   // pixel-tile base for GEMM2

    for (int j = 0; j < 4; ++j) {       // 4 paired channel chunks of 64
        // ---- load chunk weights ----
        for (int i = tid; i < 128 * 64; i += 256) {
            int r = i >> 6, k = i & 63;
            int grow = (r < 64) ? (j * 64 + r) : (HID + j * 64 + (r - 64));
            winl[i] = (h16)w_in[grow * 64 + k];
        }
        for (int i = tid; i < 128 * 9; i += 256) {
            int r = i / 9;
            int grow = (r < 64) ? (j * 64 + r) : (HID + j * 64 + (r - 64));
            wdwl[i] = w_dw[grow * 9 + (i % 9)];
        }
        for (int i = tid; i < 128; i += 256) {
            int grow = (i < 64) ? (j * 64 + i) : (HID + j * 64 + (i - 64));
            binl[i] = b_in[grow];
            bdwl[i] = b_dw[grow];
        }
        __syncthreads();

        // ---- GEMM1 (WMMA): [128,64] x [64, 336px], wave w owns Mtile w ----
        {
            int row = wave * 16 + (lane & 15);
            v16h a0 = load_A(winl, row, 64, 0,  lane);
            v16h a1 = load_A(winl, row, 64, 32, lane);
            for (int ni = 0; ni < NPAD / 16; ++ni) {
                int px = ni * 16 + (lane & 15);
                v16h b0 = load_B(xl, px, 64, 0,  lane);
                v16h b1 = load_B(xl, px, 64, 32, lane);
                v8f c = (v8f){0.f,0.f,0.f,0.f,0.f,0.f,0.f,0.f};
                c = wmma_f16f32(a0, b0, c);
                c = wmma_f16f32(a1, b1, c);
                int rbase = wave * 16 + (lane >> 4) * 8;
#pragma unroll
                for (int r = 0; r < 8; ++r) {
                    int ch = rbase + r;
                    y1l[(size_t)px * 128 + ch] = (h16)(c[r] + binl[ch]);
                }
            }
        }
        __syncthreads();

        // ---- depthwise 3x3 + exact-GELU gate -> g[px][64] f16 ----
        for (int i = tid; i < 64 * 256; i += 256) {
            int gc = i >> 8, px = i & 255;
            int py = px >> 4, pxx = px & 15;
            float av = 0.f, bv = 0.f;
#pragma unroll
            for (int ky = 0; ky < 3; ++ky)
#pragma unroll
                for (int kx = 0; kx < 3; ++kx) {
                    int gy = ty0 + py + ky - 1, gx = tx0 + pxx + kx - 1;
                    if (gy < 0 || gy >= IMGH || gx < 0 || gx >= IMGW) continue; // SAME pad of y1
                    int hn = (py + ky) * HALOW + (pxx + kx);
                    float w1 = wdwl[gc * 9 + ky * 3 + kx];
                    float w2 = wdwl[(gc + 64) * 9 + ky * 3 + kx];
                    av += w1 * (float)y1l[(size_t)hn * 128 + gc];
                    bv += w2 * (float)y1l[(size_t)hn * 128 + 64 + gc];
                }
            av += bdwl[gc];
            bv += bdwl[64 + gc];
            gl[(size_t)px * 64 + gc] = (h16)(gelu_exact(av) * bv);
        }
        __syncthreads();

        // ---- GEMM2 (WMMA) accumulate: w_out[:, chunk] x g ----
        {
            int row = mi2 * 16 + (lane & 15);
            v16h a0 = load_A(woutl, row, 256, j * 64,      lane);
            v16h a1 = load_A(woutl, row, 256, j * 64 + 32, lane);
#pragma unroll
            for (int i = 0; i < 8; ++i) {
                int px = (nbase + i) * 16 + (lane & 15);
                v16h b0 = load_B(gl, px, 64, 0,  lane);
                v16h b1 = load_B(gl, px, 64, 32, lane);
                acc[i] = wmma_f16f32(a0, b0, acc[i]);
                acc[i] = wmma_f16f32(a1, b1, acc[i]);
            }
        }
        __syncthreads();
    }

    // ---- spill conv_out tile (+bias) to LDS f32 (reuse y1 region) ----
    float* outls = (float*)(smem + OFF_Y1);
#pragma unroll
    for (int i = 0; i < 8; ++i) {
        int px    = (nbase + i) * 16 + (lane & 15);
        int rbase = mi2 * 16 + (lane >> 4) * 8;
#pragma unroll
        for (int r = 0; r < 8; ++r) {
            int ch = rbase + r;
            outls[(size_t)ch * 256 + px] = acc[i][r] + boutl[ch];
        }
    }
    __syncthreads();

    // ---- per (channel, 8x8 patch): rfft2 * mask -> irfft2, +residual ----
    {
        int ch  = tid & 63;
        int pat = tid >> 6;                 // 4 patches in a 16x16 tile
        int pu0 = (pat >> 1) * 8, pv0 = (pat & 1) * 8;
        const float* cs = trig;             // cos(2*pi*k/8)
        const float* sn = trig + 8;         // sin(2*pi*k/8)
        float Fr[8][5], Fi[8][5];
#pragma unroll
        for (int k2 = 0; k2 < 5; ++k2) {
            float Rr[8], Ri[8];
#pragma unroll
            for (int u = 0; u < 8; ++u) {
                float ar = 0.f, ai = 0.f;
#pragma clang loop unroll(disable)
                for (int v = 0; v < 8; ++v) {
                    float yv = outls[(size_t)ch * 256 + (pu0 + u) * 16 + pv0 + v];
                    int t = (k2 * v) & 7;
                    ar += yv * cs[t];
                    ai -= yv * sn[t];
                }
                Rr[u] = ar; Ri[u] = ai;
            }
#pragma unroll
            for (int k1 = 0; k1 < 8; ++k1) {
                float fr = 0.f, fi = 0.f;
#pragma unroll
                for (int u = 0; u < 8; ++u) {
                    int t = (k1 * u) & 7;
                    fr +=  Rr[u] * cs[t] + Ri[u] * sn[t];
                    fi += -Rr[u] * sn[t] + Ri[u] * cs[t];
                }
                float m = maskl[ch * 40 + k1 * 5 + k2];
                Fr[k1][k2] = fr * m;
                Fi[k1][k2] = fi * m;
            }
        }
        // inverse (Hermitian-folded) + residual + global store
        const float* xres = xb + (size_t)ch * HWPIX;
        float* wy = ws_y + ((size_t)b * DIMC + ch) * HWPIX;
#pragma clang loop unroll(disable)
        for (int u = 0; u < 8; ++u) {
            int gy = ty0 + pu0 + u;
#pragma clang loop unroll(disable)
            for (int v = 0; v < 8; ++v) {
                float s = 0.f;
#pragma unroll
                for (int k1 = 0; k1 < 8; ++k1) {
                    { int t = (k1 * u) & 7;
                      s += Fr[k1][0] * cs[t] - Fi[k1][0] * sn[t]; }
#pragma unroll
                    for (int k2 = 1; k2 < 4; ++k2) {
                        int t = (k1 * u + k2 * v) & 7;
                        s += 2.f * (Fr[k1][k2] * cs[t] - Fi[k1][k2] * sn[t]);
                    }
                    { int t = (k1 * u + 4 * v) & 7;
                      s += Fr[k1][4] * cs[t] - Fi[k1][4] * sn[t]; }
                }
                s *= (1.f / 64.f);
                int gx = tx0 + pv0 + v;
                wy[gy * IMGW + gx] = xres[gy * IMGW + gx] + s;
            }
        }
    }
}

// =====================================================================
// Kernel 2: Scharr depthwise edges -> es, per-channel sum/sumsq atomics
// =====================================================================
__global__ __launch_bounds__(256, 1)
void k_scharr(const float* __restrict__ ws_y, const float* __restrict__ swx,
              const float* __restrict__ swy, float* __restrict__ ws_es,
              float* __restrict__ stats) {
    __shared__ float red1[256];
    __shared__ float red2[256];
    const int tid = threadIdx.x;
    const int c   = tid >> 2;       // channel (0..63)
    const int q   = tid & 3;        // pixel-quarter within tile
    const int tx0 = blockIdx.x * TILE;
    const int ty0 = blockIdx.y * TILE;
    const int b   = blockIdx.z;
    const float* yb = ws_y + ((size_t)b * DIMC + c) * HWPIX;
    float* eb = ws_es + ((size_t)b * DIMC + c) * HWPIX;

    float wx[9], wy9[9];
#pragma unroll
    for (int k = 0; k < 9; ++k) { wx[k] = swx[c * 9 + k]; wy9[k] = swy[c * 9 + k]; }

    float s1 = 0.f, s2 = 0.f;
#pragma clang loop unroll(disable)
    for (int i = 0; i < 64; ++i) {
        int px = q * 64 + i;
        int gy = ty0 + (px >> 4), gx = tx0 + (px & 15);
        float ex = 0.f, ey = 0.f;
#pragma unroll
        for (int ky = 0; ky < 3; ++ky)
#pragma unroll
            for (int kx = 0; kx < 3; ++kx) {
                int yy = gy + ky - 1, xx = gx + kx - 1;
                float v = (yy >= 0 && yy < IMGH && xx >= 0 && xx < IMGW)
                              ? yb[yy * IMGW + xx] : 0.f;
                ex += wx[ky * 3 + kx] * v;
                ey += wy9[ky * 3 + kx] * v;
            }
        float es = sqrtf(ex * ex + ey * ey);
        eb[gy * IMGW + gx] = es;
        s1 += es;
        s2 += es * es;
    }
    red1[tid] = s1; red2[tid] = s2;
    __syncthreads();
    if (q == 0) {
        float t1 = red1[tid] + red1[tid + 1] + red1[tid + 2] + red1[tid + 3];
        float t2 = red2[tid] + red2[tid + 1] + red2[tid + 2] + red2[tid + 3];
        atomicAdd(&stats[c], t1);
        atomicAdd(&stats[64 + c], t2);
    }
}

// =====================================================================
// Kernel 3: BN (batch stats) + GELU attention -> out = y * att
// =====================================================================
__global__ __launch_bounds__(256, 1)
void k_norm(const float* __restrict__ ws_y, const float* __restrict__ ws_es,
            const float* __restrict__ stats, const float* __restrict__ gamma,
            const float* __restrict__ beta, float* __restrict__ out) {
    size_t i = (size_t)blockIdx.x * 256 + threadIdx.x;
    int c = (int)((i >> 16) & 63);   // layout [B][64][65536]
    float inv_n = 1.f / NPIX_PER_CH;
    float mean = stats[c] * inv_n;
    float var  = stats[64 + c] * inv_n - mean * mean;
    float es = ws_es[i];
    float n  = (es - mean) * rsqrtf(var + 1e-5f) * gamma[c] + beta[c];
    out[i] = ws_y[i] * gelu_exact(n);
}

__global__ void k_zero(float* p) { p[threadIdx.x] = 0.f; }

// =====================================================================
extern "C" void kernel_launch(void* const* d_in, const int* in_sizes, int n_in,
                              void* d_out, int out_size, void* d_ws, size_t ws_size,
                              hipStream_t stream) {
    const float* x      = (const float*)d_in[0];
    const float* w_in   = (const float*)d_in[1];
    const float* b_in   = (const float*)d_in[2];
    const float* w_dw   = (const float*)d_in[3];
    const float* b_dw   = (const float*)d_in[4];
    const float* fmask  = (const float*)d_in[5];
    const float* w_out  = (const float*)d_in[6];
    const float* b_out  = (const float*)d_in[7];
    const float* swx    = (const float*)d_in[8];
    const float* swy    = (const float*)d_in[9];
    const float* gamma  = (const float*)d_in[10];
    const float* beta   = (const float*)d_in[11];

    float* ws_y   = (float*)d_ws;
    float* ws_es  = ws_y + NTOT;
    float* stats  = ws_es + NTOT;     // 128 floats

    (void)hipFuncSetAttribute((const void*)k_edffn,
                              hipFuncAttributeMaxDynamicSharedMemorySize, K1_SMEM);

    k_zero<<<1, 128, 0, stream>>>(stats);

    dim3 gTiles(IMGW / TILE, IMGH / TILE, BATCH);   // (16,16,8)
    k_edffn<<<gTiles, 256, K1_SMEM, stream>>>(x, w_in, b_in, w_dw, b_dw, fmask,
                                              w_out, b_out, ws_y);
    k_scharr<<<gTiles, 256, 0, stream>>>(ws_y, swx, swy, ws_es, stats);

    k_norm<<<(unsigned)(NTOT / 256), 256, 0, stream>>>(ws_y, ws_es, stats,
                                                       gamma, beta, (float*)d_out);
}
